// Linear_31172872634766
// MI455X (gfx1250) — matrix operand order repaired and checked by reading, not compiled, not run on hardware
//
#include <hip/hip_runtime.h>
#include <hip/hip_bf16.h>

#define SEQ   720
#define PRED  720
#define CH    321
#define BATCH 64

#define NSTRIPS 3          // N strips per channel (grid.y)
#define WG_N    240        // output columns per workgroup
#define THREADS 384        // 12 waves (wave32)
#define TILES_PER_WAVE 5   // 16-wide N tiles per wave
#define KC      32         // K per WMMA step (bf16)
#define KCHUNKS 23         // ceil(720/32), last chunk is 16 + zero pad
#define LDA     34         // padded K stride in LDS (shorts) to avoid bank conflicts

typedef __attribute__((ext_vector_type(16))) __bf16 v16bf;
typedef __attribute__((ext_vector_type(8)))  float  v8f;

__device__ __forceinline__ unsigned short f32_to_bf16(float f) {
  unsigned int u = __float_as_uint(f);
  u += 0x7FFFu + ((u >> 16) & 1u);          // round-to-nearest-even
  return (unsigned short)(u >> 16);
}
__device__ __forceinline__ float bf16_to_f32(unsigned short hbits) {
  return __uint_as_float(((unsigned int)hbits) << 16);
}

__global__ __launch_bounds__(THREADS)
void dlinear_wmma_kernel(const float* __restrict__ x,
                         const float* __restrict__ W,
                         const float* __restrict__ bias,
                         float* __restrict__ out)
{
  // hi/lo bf16 planes for the split-precision (bf16x3) scheme
  __shared__ unsigned short Ah[BATCH * LDA];
  __shared__ unsigned short Al[BATCH * LDA];
  __shared__ unsigned short Bh[WG_N * LDA];
  __shared__ unsigned short Bl[WG_N * LDA];

  const int c     = blockIdx.x;             // channel
  const int pbase = blockIdx.y * WG_N;      // N strip base
  const int tid   = threadIdx.x;
  const int wave  = tid >> 5;
  const int lane  = tid & 31;
  const int hw    = lane >> 4;              // half-wave (0/1)
  const int ln    = lane & 15;
  const int wm    = wave & 3;               // M tile (4 x 16 = 64 batch rows)
  const int wn    = wave >> 2;              // 0..2 -> 80-wide N sub-strip
  const int mbase = wm * 16;
  const int nbase = wn * (TILES_PER_WAVE * 16);

  const float* Wc = W + (size_t)c * PRED * SEQ;

  // accumulators initialized with bias (same bias for all 64 batch rows of a column)
  v8f acc[TILES_PER_WAVE];
#pragma unroll
  for (int t = 0; t < TILES_PER_WAVE; ++t) {
    float bv = bias[(size_t)c * PRED + (pbase + nbase + t * 16 + ln)];
    acc[t] = (v8f){bv, bv, bv, bv, bv, bv, bv, bv};
  }

  for (int kc = 0; kc < KCHUNKS; ++kc) {
    const int s0 = kc * KC;
    __syncthreads();   // protect previous iteration's LDS reads

    // ---- stage W tile: WG_N rows x 32 K, coalesced float4 ----
#pragma unroll
    for (int i = 0; i < (WG_N * 8) / THREADS; ++i) {   // 5 iters
      int idx = tid + i * THREADS;                     // 0..1919
      int row = idx >> 3;
      int j   = idx & 7;
      int s   = s0 + j * 4;
      float4 w4 = {0.f, 0.f, 0.f, 0.f};
      if (s < SEQ)                                     // 720 % 4 == 0 -> all-or-nothing
        w4 = *(const float4*)(Wc + (size_t)(pbase + row) * SEQ + s);
      float vals[4] = {w4.x, w4.y, w4.z, w4.w};
      int o = row * LDA + j * 4;
#pragma unroll
      for (int e = 0; e < 4; ++e) {
        unsigned short hi = f32_to_bf16(vals[e]);
        Bh[o + e] = hi;
        Bl[o + e] = f32_to_bf16(vals[e] - bf16_to_f32(hi));
      }
    }

    // ---- stage x tile: 64 batch rows x 32 K (gather, stride C) ----
#pragma unroll
    for (int i = 0; i < 6; ++i) {
      int idx = tid + i * THREADS;
      if (idx < BATCH * KC) {
        int br = idx >> 5;
        int k  = idx & 31;
        int s  = s0 + k;
        float xv = 0.f;
        if (s < SEQ) xv = x[((size_t)br * SEQ + s) * CH + c];
        unsigned short hi = f32_to_bf16(xv);
        Ah[br * LDA + k] = hi;
        Al[br * LDA + k] = f32_to_bf16(xv - bf16_to_f32(hi));
      }
    }
    __syncthreads();

    // ---- A fragments (one M tile per wave), per CDNA5 16-bit A layout ----
    union Frag { v16bf v; unsigned int u[8]; };
    Frag ah, al;
    const unsigned int* Ah32 = (const unsigned int*)Ah;
    const unsigned int* Al32 = (const unsigned int*)Al;
    const int m = mbase + ln;
#pragma unroll
    for (int vr = 0; vr < 8; ++vr) {
      int k0  = ((vr >> 2) << 4) + (hw << 3) + ((vr & 3) << 1);
      int off = (m * LDA + k0) >> 1;     // k0 even, row stride even -> 4B aligned
      ah.u[vr] = Ah32[off];
      al.u[vr] = Al32[off];
    }

    const unsigned int* Bh32 = (const unsigned int*)Bh;
    const unsigned int* Bl32 = (const unsigned int*)Bl;
#pragma unroll
    for (int t = 0; t < TILES_PER_WAVE; ++t) {
      Frag bh, bl;
      const int ncol = nbase + t * 16 + ln;
#pragma unroll
      for (int vr = 0; vr < 8; ++vr) {
        int k0  = ((vr >> 2) << 4) + (hw << 3) + ((vr & 3) << 1);
        int off = (ncol * LDA + k0) >> 1;
        bh.u[vr] = Bh32[off];
        bl.u[vr] = Bl32[off];
      }
      // bf16x3: hi*hi + hi*lo + lo*hi, f32 accumulate
      acc[t] = __builtin_amdgcn_wmma_f32_16x16x32_bf16(false, ah.v, false, bh.v,
                                                       (short)0, acc[t], false, false);
      acc[t] = __builtin_amdgcn_wmma_f32_16x16x32_bf16(false, ah.v, false, bl.v,
                                                       (short)0, acc[t], false, false);
      acc[t] = __builtin_amdgcn_wmma_f32_16x16x32_bf16(false, al.v, false, bh.v,
                                                       (short)0, acc[t], false, false);
    }
  }

  // ---- write out [B, P, C]: out[(b*720 + p)*321 + c] ----
#pragma unroll
  for (int t = 0; t < TILES_PER_WAVE; ++t) {
    int p = pbase + nbase + t * 16 + ln;
#pragma unroll
    for (int r = 0; r < 8; ++r) {
      int brow = mbase + hw * 8 + r;
      out[((size_t)brow * PRED + p) * CH + c] = acc[t][r];
    }
  }
}

extern "C" void kernel_launch(void* const* d_in, const int* in_sizes, int n_in,
                              void* d_out, int out_size, void* d_ws, size_t ws_size,
                              hipStream_t stream) {
  const float* x    = (const float*)d_in[0];
  const float* W    = (const float*)d_in[1];
  const float* bias = (const float*)d_in[2];
  float* out        = (float*)d_out;
  dim3 grid(CH, NSTRIPS);
  dlinear_wmma_kernel<<<grid, THREADS, 0, stream>>>(x, W, bias, out);
}
